// NewtonNetEmbedding_57750130262478
// MI455X (gfx1250) — matrix operand order repaired and checked by reading, not compiled, not run on hardware
//
#include <hip/hip_runtime.h>
#include <hip/hip_bf16.h>

#define NN_DIM    128
#define NN_NBASIS 16
#define NN_CUTOFF 5.0f
#define NN_ETA    10.24f   // (NBASIS/CUTOFF)^2

typedef __attribute__((ext_vector_type(16))) __bf16 v16bf;
typedef __attribute__((ext_vector_type(8)))  float  v8f;

__device__ __forceinline__ float nn_silu(float v)
{
    // x * sigmoid(x) with fast v_rcp_f32 instead of IEEE divide
    return v * __builtin_amdgcn_rcpf(1.f + __expf(-v));
}

// ---------------------------------------------------------------- utilities

__global__ __launch_bounds__(256)
void nn_embed_kernel(const int* __restrict__ species,
                     const float* __restrict__ sW, const float* __restrict__ sb,
                     float* __restrict__ xi, int N)
{
    int idx = blockIdx.x * 256 + threadIdx.x;
    if (idx >= N * NN_DIM) return;
    int n = idx >> 7, d = idx & 127;
    xi[idx] = sW[species[n] * NN_DIM + d] + sb[d];
}

__global__ __launch_bounds__(256)
void nn_zero_kernel(float* __restrict__ p, int n)
{
    int idx = blockIdx.x * 256 + threadIdx.x;
    if (idx < n) p[idx] = 0.f;
}

// transpose a 128x128 f32 (K x N) into bf16 (N x K) for contiguous B-fragment loads
__global__ __launch_bounds__(256)
void nn_transpose_bf16_kernel(const float* __restrict__ src, __bf16* __restrict__ dst)
{
    int idx = blockIdx.x * 256 + threadIdx.x;   // idx = k*128 + n
    int k = idx >> 7, n = idx & 127;
    dst[n * NN_DIM + k] = (__bf16)src[idx];
}

// ---------------------------------------------------------------- mij kernel
// mij[e,d] = ai[src,d]*ai[dst,d]*Dij[e,d]*sw[e];  xi[src,d] += mij  (fused seg-sum)
__global__ __launch_bounds__(256)
void nn_mij_kernel(const float* __restrict__ ai,
                   const int* __restrict__ esrc, const int* __restrict__ edst,
                   const float* __restrict__ dist, const float* __restrict__ sw,
                   const float* __restrict__ radW, const float* __restrict__ radb,
                   float* __restrict__ mij, float* __restrict__ xi, int E)
{
    __shared__ float rb[2][NN_NBASIS];
    int tid = threadIdx.x;
    int ty = tid >> 7;           // 0..1  (edge within block)
    int tx = tid & 127;          // dim
    int e  = blockIdx.x * 2 + ty;
    bool ok = (e < E);
    float d = ok ? dist[e] : 0.f;
    if (tx < NN_NBASIS) {
        float center = (float)tx * (NN_CUTOFF / (float)(NN_NBASIS - 1));
        float t = d - center;
        rb[ty][tx] = __expf(-NN_ETA * t * t);
    }
    __syncthreads();
    if (!ok) return;
    float D = radb[tx];
    #pragma unroll
    for (int k = 0; k < NN_NBASIS; ++k)
        D += rb[ty][k] * radW[k * NN_DIM + tx];
    int s = esrc[e], t2 = edst[e];
    float m = ai[(size_t)s * NN_DIM + tx] * ai[(size_t)t2 * NN_DIM + tx] * D * sw[e];
    mij[(size_t)e * NN_DIM + tx] = m;
    atomicAdd(&xi[(size_t)s * NN_DIM + tx], m);
}

// ---------------------------------------------------------------- WMMA fused 2-layer MLP
// Y = silu(X @ W1 + b1) @ W2 + b2 for a 16-row tile; 8 waves, each owns 16 output cols.
// MODE 0: store Y
// MODE 1: (edge rows) atomicAdd fi[src,d,c] += Y[e,d] * Fij[e,c]
// MODE 2: (edge rows) atomicAdd phi_r[src,d] += Y[e,d] * sw[e]
// MODE 3: (node rows) xi[n,d] += -Y[n,d] * scal[n,d]
template <int MODE>
__global__ __launch_bounds__(256)
void nn_mlp2_kernel(const float* __restrict__ X, int rows,
                    const __bf16* __restrict__ W1T, const float* __restrict__ b1,
                    const __bf16* __restrict__ W2T, const float* __restrict__ b2,
                    float* __restrict__ Y,
                    const int* __restrict__ esrc,
                    const float* __restrict__ Fij,
                    float* __restrict__ fiAcc,
                    const float* __restrict__ swp,
                    float* __restrict__ phir,
                    const float* __restrict__ scal,
                    float* __restrict__ xiOut)
{
    __shared__ __bf16 ldsX[16][NN_DIM + 8];
    __shared__ __bf16 ldsH[16][NN_DIM + 8];
    __shared__ int    s_src[16];
    __shared__ float  s_F[16][3];
    __shared__ float  s_sw[16];

    const int tid   = threadIdx.x;
    const int wave  = tid >> 5;
    const int lane  = tid & 31;
    const int ln15  = lane & 15;
    const int hi    = lane >> 4;
    const int rbase = blockIdx.x * 16;

    // load X tile -> bf16 LDS
    {
        int idx = tid * 8;
        int r = idx >> 7, c = idx & 127;
        int gr = rbase + r;
        #pragma unroll
        for (int i = 0; i < 8; ++i) {
            float v = (gr < rows) ? X[(size_t)gr * NN_DIM + c + i] : 0.f;
            ldsX[r][c + i] = (__bf16)v;
        }
    }
    if (MODE == 1 && tid < 16) {
        int e = rbase + tid;
        s_src[tid]  = (e < rows) ? esrc[e] : 0;
        s_F[tid][0] = (e < rows) ? Fij[(size_t)e * 3 + 0] : 0.f;
        s_F[tid][1] = (e < rows) ? Fij[(size_t)e * 3 + 1] : 0.f;
        s_F[tid][2] = (e < rows) ? Fij[(size_t)e * 3 + 2] : 0.f;
    }
    if (MODE == 2 && tid < 16) {
        int e = rbase + tid;
        s_src[tid] = (e < rows) ? esrc[e] : 0;
        s_sw[tid]  = (e < rows) ? swp[e] : 0.f;
    }
    __syncthreads();

    const int o0   = wave * 16;
    const int coln = o0 + ln15;

    // GEMM1: hidden[16 x 16] for this wave's column slice
    v8f acc = {0.f,0.f,0.f,0.f,0.f,0.f,0.f,0.f};
    #pragma unroll
    for (int kb = 0; kb < NN_DIM; kb += 32) {
        v16bf a, b;
        const __bf16* ap = &ldsX[ln15][kb + hi * 16];
        #pragma unroll
        for (int i = 0; i < 8; ++i) { a[i] = ap[i]; a[8 + i] = ap[8 + i]; }
        const __bf16* bp = &W1T[(size_t)coln * NN_DIM + kb + hi * 16];
        #pragma unroll
        for (int i = 0; i < 16; ++i) b[i] = bp[i];
        acc = __builtin_amdgcn_wmma_f32_16x16x32_bf16(false, a, false, b,
                                                      (short)0, acc, false, false);
    }
    // bias + SiLU, stash hidden as bf16
    {
        float bias = b1[coln];
        #pragma unroll
        for (int i = 0; i < 8; ++i) {
            float v = nn_silu(acc[i] + bias);
            ldsH[i + hi * 8][coln] = (__bf16)v;
        }
    }
    __syncthreads();

    // GEMM2
    v8f acc2 = {0.f,0.f,0.f,0.f,0.f,0.f,0.f,0.f};
    #pragma unroll
    for (int kb = 0; kb < NN_DIM; kb += 32) {
        v16bf a, b;
        const __bf16* ap = &ldsH[ln15][kb + hi * 16];
        #pragma unroll
        for (int i = 0; i < 8; ++i) { a[i] = ap[i]; a[8 + i] = ap[8 + i]; }
        const __bf16* bp = &W2T[(size_t)coln * NN_DIM + kb + hi * 16];
        #pragma unroll
        for (int i = 0; i < 16; ++i) b[i] = bp[i];
        acc2 = __builtin_amdgcn_wmma_f32_16x16x32_bf16(false, a, false, b,
                                                       (short)0, acc2, false, false);
    }
    float bias2 = b2[coln];

    #pragma unroll
    for (int i = 0; i < 8; ++i) {
        int r  = i + hi * 8;          // row within tile (C/D layout: M = i + 8*hi)
        int gr = rbase + r;
        if (gr >= rows) continue;
        float v = acc2[i] + bias2;
        if (MODE == 0) {
            Y[(size_t)gr * NN_DIM + coln] = v;
        } else if (MODE == 1) {
            int s = s_src[r];
            #pragma unroll
            for (int c = 0; c < 3; ++c)
                atomicAdd(&fiAcc[((size_t)s * NN_DIM + coln) * 3 + c], v * s_F[r][c]);
        } else if (MODE == 2) {
            atomicAdd(&phir[(size_t)s_src[r] * NN_DIM + coln], v * s_sw[r]);
        } else {
            size_t idx = (size_t)gr * NN_DIM + coln;
            xiOut[idx] += -v * scal[idx];
        }
    }
}

// ---------------------------------------------------------------- F MLP (out=1) + Fij
// Fscal[e] = silu(mij@W1+b1)@W2 + b2;  Fij[e,c] = Fscal * vec[e,c]/dist[e]*sw[e]
__global__ __launch_bounds__(256)
void nn_mlpF_kernel(const float* __restrict__ X, int rows,
                    const __bf16* __restrict__ W1T, const float* __restrict__ b1,
                    const float* __restrict__ W2, const float* __restrict__ b2,
                    const float* __restrict__ dist, const float* __restrict__ vec,
                    const float* __restrict__ swp,
                    float* __restrict__ Fij)
{
    __shared__ __bf16 ldsX[16][NN_DIM + 8];
    __shared__ float  ldsHf[16][NN_DIM + 1];

    const int tid   = threadIdx.x;
    const int wave  = tid >> 5;
    const int lane  = tid & 31;
    const int ln15  = lane & 15;
    const int hi    = lane >> 4;
    const int rbase = blockIdx.x * 16;

    {
        int idx = tid * 8;
        int r = idx >> 7, c = idx & 127;
        int gr = rbase + r;
        #pragma unroll
        for (int i = 0; i < 8; ++i) {
            float v = (gr < rows) ? X[(size_t)gr * NN_DIM + c + i] : 0.f;
            ldsX[r][c + i] = (__bf16)v;
        }
    }
    __syncthreads();

    const int coln = wave * 16 + ln15;
    v8f acc = {0.f,0.f,0.f,0.f,0.f,0.f,0.f,0.f};
    #pragma unroll
    for (int kb = 0; kb < NN_DIM; kb += 32) {
        v16bf a, b;
        const __bf16* ap = &ldsX[ln15][kb + hi * 16];
        #pragma unroll
        for (int i = 0; i < 8; ++i) { a[i] = ap[i]; a[8 + i] = ap[8 + i]; }
        const __bf16* bp = &W1T[(size_t)coln * NN_DIM + kb + hi * 16];
        #pragma unroll
        for (int i = 0; i < 16; ++i) b[i] = bp[i];
        acc = __builtin_amdgcn_wmma_f32_16x16x32_bf16(false, a, false, b,
                                                      (short)0, acc, false, false);
    }
    {
        float bias = b1[coln];
        #pragma unroll
        for (int i = 0; i < 8; ++i) {
            ldsHf[i + hi * 8][coln] = nn_silu(acc[i] + bias);
        }
    }
    __syncthreads();

    if (tid < 16) {
        int e = rbase + tid;
        if (e < rows) {
            float dot = b2[0];
            #pragma unroll 4
            for (int k = 0; k < NN_DIM; ++k)
                dot += ldsHf[tid][k] * W2[k];
            float inv = swp[e] * __builtin_amdgcn_rcpf(dist[e]);
            #pragma unroll
            for (int c = 0; c < 3; ++c)
                Fij[(size_t)e * 3 + c] = dot * vec[(size_t)e * 3 + c] * inv;
        }
    }
}

// ---------------------------------------------------------------- di / scal update
__global__ __launch_bounds__(256)
void nn_update_di_kernel(const float* __restrict__ Rt, const float* __restrict__ fi,
                         const float* __restrict__ phir,
                         float* __restrict__ di, float* __restrict__ scal,
                         int n, int layer)
{
    int idx = blockIdx.x * 256 + threadIdx.x;
    if (idx >= n) return;
    float R  = Rt[idx];
    float f0 = fi[(size_t)idx * 3 + 0];
    float f1 = fi[(size_t)idx * 3 + 1];
    float f2 = fi[(size_t)idx * 3 + 2];
    float d0, d1, d2;
    if (layer == 0) {
        d0 = R * f0; d1 = R * f1; d2 = R * f2;
    } else {
        float p = phir[idx];
        d0 = p * di[(size_t)idx * 3 + 0] + R * f0;
        d1 = p * di[(size_t)idx * 3 + 1] + R * f1;
        d2 = p * di[(size_t)idx * 3 + 2] + R * f2;
    }
    di[(size_t)idx * 3 + 0] = d0;
    di[(size_t)idx * 3 + 1] = d1;
    di[(size_t)idx * 3 + 2] = d2;
    scal[idx] = f0 * d0 + f1 * d1 + f2 * d2;
}

// ---------------------------------------------------------------- host driver

extern "C" void kernel_launch(void* const* d_in, const int* in_sizes, int n_in,
                              void* d_out, int out_size, void* d_ws, size_t ws_size,
                              hipStream_t stream)
{
    (void)n_in; (void)out_size; (void)ws_size;
    const int*   species = (const int*)  d_in[0];
    const int*   esrc    = (const int*)  d_in[1];
    const int*   edst    = (const int*)  d_in[2];
    const float* dist    = (const float*)d_in[3];
    const float* vec     = (const float*)d_in[4];
    const float* sw      = (const float*)d_in[5];
    const float* sW      = (const float*)d_in[6];
    const float* sb      = (const float*)d_in[7];
    const float* radW    = (const float*)d_in[8];
    const float* radb    = (const float*)d_in[9];
    const float *W1[6], *B1[6], *W2[6], *B2[6];   // 0=a,1=F,2=f,3=R,4=r,5=u
    for (int m = 0; m < 6; ++m) {
        W1[m] = (const float*)d_in[10 + m * 4 + 0];
        B1[m] = (const float*)d_in[10 + m * 4 + 1];
        W2[m] = (const float*)d_in[10 + m * 4 + 2];
        B2[m] = (const float*)d_in[10 + m * 4 + 3];
    }
    const int N = in_sizes[0];
    const int E = in_sizes[1];
    float* xi = (float*)d_out;

    size_t off = 0;
    auto alloc = [&](size_t bytes) -> char* {
        off = (off + 255) & ~(size_t)255;
        char* p = (char*)d_ws + off;
        off += bytes;
        return p;
    };
    float*  ai   = (float*) alloc((size_t)N * NN_DIM * 4);
    float*  mij  = (float*) alloc((size_t)E * NN_DIM * 4);
    float*  fi   = (float*) alloc((size_t)N * NN_DIM * 3 * 4);
    float*  di   = (float*) alloc((size_t)N * NN_DIM * 3 * 4);
    float*  Rt   = (float*) alloc((size_t)N * NN_DIM * 4);
    float*  scal = (float*) alloc((size_t)N * NN_DIM * 4);
    float*  phir = (float*) alloc((size_t)N * NN_DIM * 4);
    float*  Fij  = (float*) alloc((size_t)E * 3 * 4);
    __bf16* wbf  = (__bf16*)alloc((size_t)3 * 11 * NN_DIM * NN_DIM * 2);

    auto wt = [&](int l, int slot) -> __bf16* {
        return wbf + ((size_t)l * 11 + slot) * (NN_DIM * NN_DIM);
    };
    // slots: 0 aW1, 1 aW2, 2 FW1, 3 fW1, 4 fW2, 5 RW1, 6 RW2, 7 rW1, 8 rW2, 9 uW1, 10 uW2
    for (int l = 0; l < 3; ++l) {
        const float* srcs[11] = {
            W1[0] + l * 16384, W2[0] + l * 16384, W1[1] + l * 16384,
            W1[2] + l * 16384, W2[2] + l * 16384, W1[3] + l * 16384, W2[3] + l * 16384,
            W1[4] + l * 16384, W2[4] + l * 16384, W1[5] + l * 16384, W2[5] + l * 16384 };
        for (int m = 0; m < 11; ++m)
            nn_transpose_bf16_kernel<<<64, 256, 0, stream>>>(srcs[m], wt(l, m));
    }

    nn_embed_kernel<<<(N * NN_DIM + 255) / 256, 256, 0, stream>>>(species, sW, sb, xi, N);
    nn_zero_kernel<<<(N * NN_DIM * 3 + 255) / 256, 256, 0, stream>>>(fi, N * NN_DIM * 3);

    const int nodeBlocks = (N + 15) / 16;
    const int edgeBlocks = (E + 15) / 16;
    for (int l = 0; l < 3; ++l) {
        // ai = MLP_a(xi)
        nn_mlp2_kernel<0><<<nodeBlocks, 256, 0, stream>>>(
            xi, N, wt(l, 0), B1[0] + l * NN_DIM, wt(l, 1), B2[0] + l * NN_DIM, ai,
            nullptr, nullptr, nullptr, nullptr, nullptr, nullptr, nullptr);
        // mij + fused xi segment-sum
        nn_mij_kernel<<<(E + 1) / 2, 256, 0, stream>>>(
            ai, esrc, edst, dist, sw,
            radW + (size_t)l * NN_NBASIS * NN_DIM, radb + l * NN_DIM, mij, xi, E);
        // Fij = MLP_F(mij) * dirij
        nn_mlpF_kernel<<<edgeBlocks, 256, 0, stream>>>(
            mij, E, wt(l, 2), B1[1] + l * NN_DIM, W2[1] + l * NN_DIM, B2[1] + l,
            dist, vec, sw, Fij);
        // fi += seg_sum( MLP_f(mij) x Fij )   (scatter fused into WMMA epilogue)
        nn_mlp2_kernel<1><<<edgeBlocks, 256, 0, stream>>>(
            mij, E, wt(l, 3), B1[2] + l * NN_DIM, wt(l, 4), B2[2] + l * NN_DIM, nullptr,
            esrc, Fij, fi, nullptr, nullptr, nullptr, nullptr);
        // Rt = MLP_R(xi)
        nn_mlp2_kernel<0><<<nodeBlocks, 256, 0, stream>>>(
            xi, N, wt(l, 5), B1[3] + l * NN_DIM, wt(l, 6), B2[3] + l * NN_DIM, Rt,
            nullptr, nullptr, nullptr, nullptr, nullptr, nullptr, nullptr);
        if (l > 0) {
            nn_zero_kernel<<<(N * NN_DIM + 255) / 256, 256, 0, stream>>>(phir, N * NN_DIM);
            // phi_r = seg_sum( MLP_r(mij) * sw )   (scatter fused)
            nn_mlp2_kernel<2><<<edgeBlocks, 256, 0, stream>>>(
                mij, E, wt(l, 7), B1[4] + l * NN_DIM, wt(l, 8), B2[4] + l * NN_DIM, nullptr,
                esrc, nullptr, nullptr, sw, phir, nullptr, nullptr);
        }
        nn_update_di_kernel<<<(N * NN_DIM + 255) / 256, 256, 0, stream>>>(
            Rt, fi, phir, di, scal, N * NN_DIM, l);
        // xi += -MLP_u(xi) * scal   (apply fused)
        nn_mlp2_kernel<3><<<nodeBlocks, 256, 0, stream>>>(
            xi, N, wt(l, 9), B1[5] + l * NN_DIM, wt(l, 10), B2[5] + l * NN_DIM, nullptr,
            nullptr, nullptr, nullptr, nullptr, nullptr, scal, xi);
    }
}
